// NAIS_model_65712999629183
// MI455X (gfx1250) — matrix operand order repaired and checked by reading, not compiled, not run on hardware
//
#include <hip/hip_runtime.h>
#include <hip/hip_bf16.h>

typedef __attribute__((ext_vector_type(16))) _Float16 v16h;
typedef __attribute__((ext_vector_type(8)))  _Float16 v8h;
typedef __attribute__((ext_vector_type(8)))  float    v8f;

#define NAIS_H  200
#define NAIS_HP 208          // 13 tiles of 16 rows
#define NAIS_F  64
#define NAIS_W  32
#define AS      72           // LDS row stride in halves (16B aligned, bank-spread)
#define WS      72
#define H2S     36           // f32 row stride for layer-2 output

union AFrag { v16h v; v8h h2[2]; _Float16 h[16]; };
union CFrag { v8f  v; float f[8]; };

__global__ __launch_bounds__(256) void nais_kernel(
    const int*   __restrict__ user,      const int*   __restrict__ item,
    const int*   __restrict__ hist_items,const float* __restrict__ hist_lens,
    const float* __restrict__ hist_mask,
    const float* __restrict__ Gi,        const float* __restrict__ Gj,
    const float* __restrict__ Bu,        const float* __restrict__ Bi,
    const float* __restrict__ W1,        const float* __restrict__ b1,
    const float* __restrict__ W2,        const float* __restrict__ b2,
    const float* __restrict__ W3,        const float* __restrict__ b3,
    float* __restrict__ out)
{
    __shared__ __align__(16) _Float16 sA  [NAIS_HP * AS];  // mlp_in (f16); reused as sH2 (f32)
    __shared__ __align__(16) _Float16 sH1 [NAIS_HP * AS];  // relu output (f16)
    __shared__ __align__(16) _Float16 sW1t[NAIS_F  * WS];  // W1^T  [n][k]
    __shared__ __align__(16) _Float16 sW2t[NAIS_W  * WS];  // W2^T  [n][k]
    __shared__ float sTgt[NAIS_F];
    __shared__ float sB1 [NAIS_F];
    __shared__ float sB2 [NAIS_W];
    __shared__ float sW3 [NAIS_W];
    __shared__ float sE  [256];
    __shared__ float sES [256];

    const int tid = threadIdx.x;
    const int b   = blockIdx.x;
    const int u   = user[b];
    const int it  = item[b];

    // ---------- phase 0: stage target / weights (f16 transposed) ----------
    if (tid < NAIS_F) { sTgt[tid] = Gj[(size_t)it * NAIS_F + tid]; sB1[tid] = b1[tid]; }
    if (tid < NAIS_W) { sB2[tid] = b2[tid]; sW3[tid] = W3[tid]; }
    for (int idx = tid; idx < NAIS_F * NAIS_F; idx += 256) {
        int k = idx >> 6, n = idx & 63;
        sW1t[n * WS + k] = (_Float16)W1[idx];
    }
    for (int idx = tid; idx < NAIS_F * NAIS_W; idx += 256) {
        int k = idx >> 5, n = idx & 31;
        sW2t[n * WS + k] = (_Float16)W2[idx];
    }
    __syncthreads();

    // ---------- phase 1: gather history rows, mlp_in (f16) + similarity ----------
    float sim = 0.0f, msk = 0.0f;
    if (tid < NAIS_HP) {
        if (tid < NAIS_H) {
            const int hid = hist_items[(size_t)u * NAIS_H + tid];
            msk = hist_mask[(size_t)u * NAIS_H + tid];
            const float4* grow = (const float4*)(Gi + (size_t)hid * NAIS_F);
            #pragma unroll 4
            for (int q = 0; q < NAIS_F / 4; ++q) {
                float4 g = grow[q];
                float t0 = sTgt[4*q+0], t1 = sTgt[4*q+1];
                float t2 = sTgt[4*q+2], t3 = sTgt[4*q+3];
                float p0 = g.x*t0, p1 = g.y*t1, p2 = g.z*t2, p3 = g.w*t3;
                sim += (p0 + p1) + (p2 + p3);
                _Float16* dst = &sA[tid * AS + 4*q];
                dst[0] = (_Float16)p0; dst[1] = (_Float16)p1;
                dst[2] = (_Float16)p2; dst[3] = (_Float16)p3;
            }
        } else {
            for (int q = 0; q < NAIS_F; ++q) sA[tid * AS + q] = (_Float16)0.0f;
        }
    }
    __syncthreads();

    const int wid   = tid >> 5;
    const int lane  = tid & 31;
    const int lhalf = (lane < 16) ? 0 : 1;   // which half-wave
    const int lm    = lane & 15;

    // ---------- layer 1: [208x64] = relu(mlp_in @ W1 + b1) via WMMA ----------
    for (int t = wid; t < NAIS_HP / 16; t += 8) {
        const int m = t * 16 + lm;
        AFrag a0, a1;
        // K-tile 0 (K 0..31) and K-tile 1 (K 32..63), CDNA5 16-bit A layout
        a0.h2[0] = *(const v8h*)&sA[m*AS +  0 + lhalf*8];
        a0.h2[1] = *(const v8h*)&sA[m*AS + 16 + lhalf*8];
        a1.h2[0] = *(const v8h*)&sA[m*AS + 32 + lhalf*8];
        a1.h2[1] = *(const v8h*)&sA[m*AS + 48 + lhalf*8];
        #pragma unroll
        for (int nt = 0; nt < 4; ++nt) {
            const int nb = nt * 16 + lm;
            AFrag bf0, bf1;  // B layout: lane = N, lanes 0-15 K 0..7,16..23, 16-31 K 8..15,24..31
            bf0.h2[0] = *(const v8h*)&sW1t[nb*WS +      lhalf*8];
            bf0.h2[1] = *(const v8h*)&sW1t[nb*WS + 16 + lhalf*8];
            bf1.h2[0] = *(const v8h*)&sW1t[nb*WS + 32 + lhalf*8];
            bf1.h2[1] = *(const v8h*)&sW1t[nb*WS + 48 + lhalf*8];
            CFrag acc = {};
            acc.v = __builtin_amdgcn_wmma_f32_16x16x32_f16(false, a0.v, false, bf0.v,
                                                           (short)0, acc.v, false, false);
            acc.v = __builtin_amdgcn_wmma_f32_16x16x32_f16(false, a1.v, false, bf1.v,
                                                           (short)0, acc.v, false, false);
            const int   col  = nt * 16 + lm;
            const float bias = sB1[col];
            #pragma unroll
            for (int r = 0; r < 8; ++r) {
                int row = t * 16 + r + lhalf * 8;
                sH1[row*AS + col] = (_Float16)fmaxf(acc.f[r] + bias, 0.0f);
            }
        }
    }
    __syncthreads();

    // ---------- layer 2: [208x32] = H1 @ W2 + b2 via WMMA (f32 out, alias sA) ----------
    float* sH2 = (float*)sA;
    for (int t = wid; t < NAIS_HP / 16; t += 8) {
        const int m = t * 16 + lm;
        AFrag a0, a1;
        a0.h2[0] = *(const v8h*)&sH1[m*AS +  0 + lhalf*8];
        a0.h2[1] = *(const v8h*)&sH1[m*AS + 16 + lhalf*8];
        a1.h2[0] = *(const v8h*)&sH1[m*AS + 32 + lhalf*8];
        a1.h2[1] = *(const v8h*)&sH1[m*AS + 48 + lhalf*8];
        #pragma unroll
        for (int nt = 0; nt < 2; ++nt) {
            const int nb = nt * 16 + lm;
            AFrag bf0, bf1;
            bf0.h2[0] = *(const v8h*)&sW2t[nb*WS +      lhalf*8];
            bf0.h2[1] = *(const v8h*)&sW2t[nb*WS + 16 + lhalf*8];
            bf1.h2[0] = *(const v8h*)&sW2t[nb*WS + 32 + lhalf*8];
            bf1.h2[1] = *(const v8h*)&sW2t[nb*WS + 48 + lhalf*8];
            CFrag acc = {};
            acc.v = __builtin_amdgcn_wmma_f32_16x16x32_f16(false, a0.v, false, bf0.v,
                                                           (short)0, acc.v, false, false);
            acc.v = __builtin_amdgcn_wmma_f32_16x16x32_f16(false, a1.v, false, bf1.v,
                                                           (short)0, acc.v, false, false);
            const int   col  = nt * 16 + lm;
            const float bias = sB2[col];
            #pragma unroll
            for (int r = 0; r < 8; ++r) {
                int row = t * 16 + r + lhalf * 8;
                sH2[row*H2S + col] = acc.f[r] + bias;
            }
        }
    }
    __syncthreads();

    // ---------- layer 3 + masked power-softmax numerators ----------
    float e = 0.0f, es = 0.0f;
    if (tid < NAIS_H) {
        float logit = b3[0];
        #pragma unroll
        for (int w = 0; w < NAIS_W; ++w) logit += sH2[tid*H2S + w] * sW3[w];
        e  = msk * expf(logit);
        es = e * sim;
    }
    sE[tid] = e;
    sES[tid] = es;
    __syncthreads();

    // ---------- block reduction ----------
    #pragma unroll
    for (int s = 128; s > 0; s >>= 1) {
        if (tid < s) { sE[tid] += sE[tid + s]; sES[tid] += sES[tid + s]; }
        __syncthreads();
    }

    if (tid == 0) {
        float esum  = sE[0];             // sum of masked exp(logits)
        float psum  = sES[0];            // sum exp*similarity
        float inum  = hist_lens[u];
        // weights = e / esum^0.5 ; prod = inum^-0.5 * sum(w*sim)
        float prod  = rsqrtf(inum) * psum * rsqrtf(esum);
        float x     = prod + Bu[u] + Bi[it];
        out[b] = 1.0f / (1.0f + expf(-x));
    }
}

extern "C" void kernel_launch(void* const* d_in, const int* in_sizes, int n_in,
                              void* d_out, int out_size, void* d_ws, size_t ws_size,
                              hipStream_t stream) {
    const int*   user       = (const int*)  d_in[0];
    const int*   item       = (const int*)  d_in[1];
    const int*   hist_items = (const int*)  d_in[2];
    const float* hist_lens  = (const float*)d_in[3];
    const float* hist_mask  = (const float*)d_in[4];
    const float* Gi         = (const float*)d_in[5];
    const float* Gj         = (const float*)d_in[6];
    const float* Bu         = (const float*)d_in[7];
    const float* Bi         = (const float*)d_in[8];
    const float* W1         = (const float*)d_in[9];
    const float* b1         = (const float*)d_in[10];
    const float* W2         = (const float*)d_in[11];
    const float* b2         = (const float*)d_in[12];
    const float* W3         = (const float*)d_in[13];
    const float* b3         = (const float*)d_in[14];
    float* out = (float*)d_out;

    const int B = in_sizes[0];   // batch (4096)
    nais_kernel<<<B, 256, 0, stream>>>(user, item, hist_items, hist_lens, hist_mask,
                                       Gi, Gj, Bu, Bi, W1, b1, W2, b2, W3, b3, out);
}
